// SimpleSSM_35923106464034
// MI455X (gfx1250) — matrix operand order repaired and checked by reading, not compiled, not run on hardware
//
#include <hip/hip_runtime.h>
#include <cstdint>

#define DI static __device__ __forceinline__

constexpr int Vv = 32000;
constexpr int Dd = 512;
constexpr int Ll = 2;
constexpr int Bb = 8;
constexpr int Ss = 4096;
constexpr int NROWS = Bb * Ss;      // 32768 token rows
constexpr int NC = 64;              // scan chunks per sequence
constexpr int LC = Ss / NC;         // 64 steps per chunk
constexpr float EPS = 1e-5f;

typedef __bf16 bf16;
typedef __attribute__((ext_vector_type(16))) __bf16 v16bf;
typedef __attribute__((ext_vector_type(8)))  __bf16 v8bf;
typedef __attribute__((ext_vector_type(8)))  float  v8f;
typedef __attribute__((ext_vector_type(4)))  float  v4f;

DI v8f wmma_bf16(v16bf a, v16bf b, v8f c) {
  // D(f32 16x16) = A(bf16 16x32) * B(bf16 32x16) + C
  return __builtin_amdgcn_wmma_f32_16x16x32_bf16(false, a, false, b, (short)0, c,
                                                 false, false);
}
DI v16bf cat8(v8bf lo, v8bf hi) {
  return __builtin_shufflevector(lo, hi, 0,1,2,3,4,5,6,7,8,9,10,11,12,13,14,15);
}

// CDNA5 async copy: global -> LDS, 16B per lane, tracked by ASYNCcnt.
DI void async_g2l_b128(uint32_t lds_off, const void* gaddr) {
  asm volatile("global_load_async_to_lds_b128 %0, %1, off"
               :: "v"(lds_off), "v"(gaddr) : "memory");
}
DI void wait_asynccnt0() {
  asm volatile("s_wait_asynccnt 0x0" ::: "memory");
}
DI uint32_t lds_off_of(const void* shared_ptr) {
  return (uint32_t)(uintptr_t)shared_ptr;   // addr[31:0] == LDS offset (ISA 10.2)
}

// ---------------- embedding gather: h[row,:] = emb[x[row],:] ----------------
__global__ void k_embed(const int* __restrict__ x, const float* __restrict__ emb,
                        float* __restrict__ h) {
  int tid = blockIdx.x * blockDim.x + threadIdx.x;   // NROWS * 128
  int row = tid >> 7;
  int c4  = tid & 127;
  const v4f* src = (const v4f*)(emb + (size_t)x[row] * Dd);
  v4f* dst = (v4f*)(h + (size_t)row * Dd);
  dst[c4] = src[c4];
}

// --------- per-layer scalars: a_bar, scale (bilinear discretization) --------
__global__ void k_params(const float* __restrict__ a_log,
                         const float* __restrict__ dt_log,
                         float* __restrict__ a_bar, float* __restrict__ scale) {
  int i = blockIdx.x * blockDim.x + threadIdx.x;     // L*D
  if (i >= Ll * Dd) return;
  float a    = -__expf(a_log[i]);
  float dt   = log1pf(__expf(dt_log[i])) + 1e-4f;    // softplus + eps
  float half = 0.5f * dt * a;
  a_bar[i] = (1.f + half) / (1.f - half);
  scale[i] = dt / (1.f - half);
}

// --------------- weight conversions to bf16 (b_bar pre-scaled) --------------
__global__ void k_cvt(const float* __restrict__ b_mat, const float* __restrict__ c_mat,
                      const float* __restrict__ d_w, const float* __restrict__ scale,
                      bf16* __restrict__ wb, bf16* __restrict__ wc, bf16* __restrict__ wd) {
  size_t i = (size_t)blockIdx.x * blockDim.x + threadIdx.x;  // L*D*D
  int l = (int)(i / ((size_t)Dd * Dd));
  int e = (int)((i / Dd) % Dd);
  wb[i] = (bf16)(scale[l * Dd + e] * b_mat[i]);
  wc[i] = (bf16)c_mat[i];
  wd[i] = (bf16)d_w[i];
}

// --------------------- LayerNorm, one wave32 per row ------------------------
__global__ void k_ln(const float* __restrict__ h, const float* __restrict__ w,
                     const float* __restrict__ b, bf16* __restrict__ u) {
  int wave = threadIdx.x >> 5;
  int lane = threadIdx.x & 31;
  int row  = blockIdx.x * (blockDim.x >> 5) + wave;
  const float* hr = h + (size_t)row * Dd;
  float xv[16];
  float s = 0.f;
  #pragma unroll
  for (int i = 0; i < 16; ++i) { xv[i] = hr[i * 32 + lane]; s += xv[i]; }
  #pragma unroll
  for (int m = 16; m >= 1; m >>= 1) s += __shfl_xor(s, m, 32);
  float mu = s * (1.f / Dd);
  float var = 0.f;
  #pragma unroll
  for (int i = 0; i < 16; ++i) { float d = xv[i] - mu; var += d * d; }
  #pragma unroll
  for (int m = 16; m >= 1; m >>= 1) var += __shfl_xor(var, m, 32);
  float rstd = rsqrtf(var * (1.f / Dd) + EPS);
  bf16* ur = u + (size_t)row * Dd;
  #pragma unroll
  for (int i = 0; i < 16; ++i) {
    int c = i * 32 + lane;
    ur[c] = (bf16)((xv[i] - mu) * rstd * w[c] + b[c]);
  }
}

// ---------------- GEMM1: v[N,512] = u(bf16) @ b_bar^T (WMMA) ----------------
// Block = 8 waves sharing one 64-col B panel async-staged into LDS.
// Each wave owns a 16(M) x 64(N) tile; BOTH the A fragment (global) and the
// 4 B fragments (LDS) are double-buffered across K so each WMMA consumes
// registers loaded one iteration earlier (no per-WMMA dscnt stalls).
__global__ void k_gemm_v(const bf16* __restrict__ A, const bf16* __restrict__ W,
                         float* __restrict__ out) {
  __shared__ alignas(32) bf16 sB[64 * Dd];            // 64 KB weight panel
  const int lane = threadIdx.x & 31;
  const int wave = threadIdx.x >> 5;
  const int nBlk  = blockIdx.x >> 8;                  // 0..7
  const int mTile = ((blockIdx.x & 255) << 3) + wave; // 0..2047
  const int row0 = mTile << 4;
  const int n0   = nBlk << 6;
  const int m = lane & 15, g = lane >> 4;

  // ---- async-stage the contiguous 64KB panel W[n0..n0+63][*] into LDS ----
  {
    const char* gB = (const char*)(W + (size_t)n0 * Dd);
    const char* lB = (const char*)sB;
    #pragma unroll
    for (int i = 0; i < 16; ++i) {
      int off = (i * 256 + (int)threadIdx.x) * 16;
      async_g2l_b128(lds_off_of(lB + off), gB + off);
    }
    wait_asynccnt0();
    __syncthreads();
  }

  v8f acc0 = {}, acc1 = {}, acc2 = {}, acc3 = {};
  const bf16* pa  = A + (size_t)(row0 + m) * Dd + g * 16;  // A: lane holds row m
  const bf16* pbl = sB + (size_t)m * Dd + g * 16;          // B: lane holds col n
  v16bf a_cur = cat8(*(const v8bf*)pa, *(const v8bf*)(pa + 8));
  v16bf b_cur[4], b_nxt[4];
  #pragma unroll
  for (int s = 0; s < 4; ++s) b_cur[s] = *(const v16bf*)(pbl + s * 16 * Dd);

  #pragma unroll 4
  for (int k = 0; k < Dd; k += 32) {
    v16bf a_nxt = a_cur;
    if (k + 32 < Dd) {
      // issue next-step LDS + global loads BEFORE this step's WMMAs
      #pragma unroll
      for (int s = 0; s < 4; ++s)
        b_nxt[s] = *(const v16bf*)(pbl + s * 16 * Dd + k + 32);
      const bf16* pan = pa + 32;
      a_nxt = cat8(*(const v8bf*)pan, *(const v8bf*)(pan + 8));
      __builtin_prefetch(pan + 32, 0, 1);
    } else {
      #pragma unroll
      for (int s = 0; s < 4; ++s) b_nxt[s] = b_cur[s];
    }
    acc0 = wmma_bf16(a_cur, b_cur[0], acc0);
    acc1 = wmma_bf16(a_cur, b_cur[1], acc1);
    acc2 = wmma_bf16(a_cur, b_cur[2], acc2);
    acc3 = wmma_bf16(a_cur, b_cur[3], acc3);
    a_cur = a_nxt;
    #pragma unroll
    for (int s = 0; s < 4; ++s) b_cur[s] = b_nxt[s];
    pa += 32;
  }
  float* po = out + (size_t)row0 * Dd + n0;
  #pragma unroll
  for (int r = 0; r < 8; ++r) {                  // C layout: VGPR r -> rows r, r+8
    size_t off = (size_t)(r + 8 * g) * Dd + m;
    po[off]      = acc0[r];
    po[off + 16] = acc1[r];
    po[off + 32] = acc2[r];
    po[off + 48] = acc3[r];
  }
}

// ------------- chunked parallel scan: pass 1 (local scan in-place) ----------
__global__ void k_scan_local(float* __restrict__ v, const float* __restrict__ a_bar,
                             float* __restrict__ last) {
  int tid = blockIdx.x * blockDim.x + threadIdx.x;  // Bb*NC*Dd
  int d = tid & (Dd - 1);
  int r = tid >> 9;                                 // (b, chunk)
  int c = r & (NC - 1);
  int b = r >> 6;
  float a = a_bar[d];
  size_t base = (size_t)(b * Ss + c * LC) * Dd + d;
  float y = 0.f;
  for (int t = 0; t < LC; ++t) {
    size_t idx = base + (size_t)t * Dd;
    y = fmaf(a, y, v[idx]);
    v[idx] = y;
  }
  last[(size_t)r * Dd + d] = y;
}

// ---- pass 2: serial carry sweep across chunks (in-place last -> carry-in) --
__global__ void k_scan_carry(float* __restrict__ cl, const float* __restrict__ a_bar) {
  int tid = blockIdx.x * blockDim.x + threadIdx.x;  // Bb*Dd channels
  int d = tid & (Dd - 1);
  int b = tid >> 9;
  float a = a_bar[d];
  float ap = a;
  #pragma unroll
  for (int i = 0; i < 6; ++i) ap *= ap;             // a^64 = a^LC
  float sIn = 0.f;
  for (int c = 0; c < NC; ++c) {
    size_t idx = (size_t)(b * NC + c) * Dd + d;
    float lastv = cl[idx];
    cl[idx] = sIn;                                  // incoming carry for chunk c
    sIn = fmaf(ap, sIn, lastv);
  }
}

// -------- pass 3: fix-up states += a^(t+1)*carry, emit bf16 states ----------
__global__ void k_scan_fix(const float* __restrict__ v, const float* __restrict__ cl,
                           const float* __restrict__ a_bar, bf16* __restrict__ st) {
  int tid = blockIdx.x * blockDim.x + threadIdx.x;  // Bb*NC*Dd
  int d = tid & (Dd - 1);
  int r = tid >> 9;
  int c = r & (NC - 1);
  int b = r >> 6;
  float a = a_bar[d];
  float s = cl[(size_t)r * Dd + d];
  size_t base = (size_t)(b * Ss + c * LC) * Dd + d;
  float p = a;
  for (int t = 0; t < LC; ++t) {
    size_t idx = base + (size_t)t * Dd;
    st[idx] = (bf16)(v[idx] + p * s);
    p *= a;
  }
}

// ------ GEMM2 fused: h += states@c^T + u@d^T + d_b (two WMMA chains) --------
// Both weight panels async-staged into LDS; two A streams double-buffered;
// all 8 B fragments batch-loaded per K-step (one dscnt wait per 8 WMMAs).
__global__ void k_gemm_y(const bf16* __restrict__ Ast, const bf16* __restrict__ Au,
                         const bf16* __restrict__ Wc, const bf16* __restrict__ Wd,
                         const float* __restrict__ db, float* __restrict__ h) {
  __shared__ alignas(32) bf16 sC[64 * Dd];            // 64 KB c panel
  __shared__ alignas(32) bf16 sD[64 * Dd];            // 64 KB d panel
  const int lane = threadIdx.x & 31;
  const int wave = threadIdx.x >> 5;
  const int nBlk  = blockIdx.x >> 8;
  const int mTile = ((blockIdx.x & 255) << 3) + wave;
  const int row0 = mTile << 4;
  const int n0   = nBlk << 6;
  const int m = lane & 15, g = lane >> 4;

  {
    const char* gC = (const char*)(Wc + (size_t)n0 * Dd);
    const char* gD = (const char*)(Wd + (size_t)n0 * Dd);
    const char* lC = (const char*)sC;
    const char* lD = (const char*)sD;
    #pragma unroll
    for (int i = 0; i < 16; ++i) {
      int off = (i * 256 + (int)threadIdx.x) * 16;
      async_g2l_b128(lds_off_of(lC + off), gC + off);
      async_g2l_b128(lds_off_of(lD + off), gD + off);
    }
    wait_asynccnt0();
    __syncthreads();
  }

  v8f acc[4] = {};
  const bf16* pas = Ast + (size_t)(row0 + m) * Dd + g * 16;
  const bf16* pau = Au  + (size_t)(row0 + m) * Dd + g * 16;
  const bf16* pcl = sC + (size_t)m * Dd + g * 16;
  const bf16* pdl = sD + (size_t)m * Dd + g * 16;
  v16bf as_cur = cat8(*(const v8bf*)pas, *(const v8bf*)(pas + 8));
  v16bf au_cur = cat8(*(const v8bf*)pau, *(const v8bf*)(pau + 8));
  #pragma unroll 2
  for (int k = 0; k < Dd; k += 32) {
    // batch all 16 LDS loads for this step, then the next-step global loads,
    // then run the 8-WMMA chain behind a single dscnt wait
    v16bf bc[4], bd[4];
    #pragma unroll
    for (int sub = 0; sub < 4; ++sub) {
      bc[sub] = *(const v16bf*)(pcl + (size_t)sub * 16 * Dd + k);
      bd[sub] = *(const v16bf*)(pdl + (size_t)sub * 16 * Dd + k);
    }
    v16bf as_nxt = as_cur, au_nxt = au_cur;
    if (k + 32 < Dd) {
      const bf16* ps = pas + 32;
      const bf16* pu = pau + 32;
      as_nxt = cat8(*(const v8bf*)ps, *(const v8bf*)(ps + 8));
      au_nxt = cat8(*(const v8bf*)pu, *(const v8bf*)(pu + 8));
      __builtin_prefetch(ps + 32, 0, 1);
      __builtin_prefetch(pu + 32, 0, 1);
    }
    #pragma unroll
    for (int sub = 0; sub < 4; ++sub) {
      acc[sub] = wmma_bf16(as_cur, bc[sub], acc[sub]);
      acc[sub] = wmma_bf16(au_cur, bd[sub], acc[sub]);
    }
    as_cur = as_nxt; au_cur = au_nxt;
    pas += 32; pau += 32;
  }
  float* ph = h + (size_t)row0 * Dd + n0;
  #pragma unroll
  for (int r = 0; r < 8; ++r) {
    size_t off = (size_t)(r + 8 * g) * Dd + m;
    #pragma unroll
    for (int sub = 0; sub < 4; ++sub)
      ph[off + sub * 16] += acc[sub][r] + db[n0 + sub * 16 + m];
  }
}

// ------- final LayerNorm of last token into padded 16x512 bf16 tile ---------
__global__ void k_final_ln(const float* __restrict__ h, const float* __restrict__ w,
                           const float* __restrict__ b, bf16* __restrict__ hf) {
  int wave = threadIdx.x >> 5;   // 16 waves, rows 0..15 (8..15 are zero pad)
  int lane = threadIdx.x & 31;
  bf16* out = hf + wave * Dd;
  if (wave >= Bb) {
    #pragma unroll
    for (int i = 0; i < 16; ++i) out[i * 32 + lane] = (bf16)0.f;
    return;
  }
  const float* hr = h + (size_t)(wave * Ss + (Ss - 1)) * Dd;
  float xv[16];
  float s = 0.f;
  #pragma unroll
  for (int i = 0; i < 16; ++i) { xv[i] = hr[i * 32 + lane]; s += xv[i]; }
  #pragma unroll
  for (int m = 16; m >= 1; m >>= 1) s += __shfl_xor(s, m, 32);
  float mu = s * (1.f / Dd);
  float var = 0.f;
  #pragma unroll
  for (int i = 0; i < 16; ++i) { float d = xv[i] - mu; var += d * d; }
  #pragma unroll
  for (int m = 16; m >= 1; m >>= 1) var += __shfl_xor(var, m, 32);
  float rstd = rsqrtf(var * (1.f / Dd) + EPS);
  #pragma unroll
  for (int i = 0; i < 16; ++i) {
    int c = i * 32 + lane;
    out[c] = (bf16)((xv[i] - mu) * rstd * w[c] + b[c]);
  }
}

// ---- output projection: logits[8,32000] = hf @ out_w^T + out_b (WMMA) ------
__global__ void k_out(const bf16* __restrict__ hf, const float* __restrict__ Wo,
                      const float* __restrict__ bo, float* __restrict__ out) {
  const int lane = threadIdx.x & 31;
  const int wid  = blockIdx.x * (blockDim.x >> 5) + (threadIdx.x >> 5);
  if (wid >= Vv / 64) return;                      // 500 wave tiles of 64 cols
  const int n0 = wid << 6;
  const int m = lane & 15, g = lane >> 4;
  v8f acc[4] = {};
  const bf16*  pa = hf + m * Dd + g * 16;
  const float* pw = Wo + (size_t)(n0 + m) * Dd + g * 16;
  for (int k = 0; k < Dd; k += 32) {
    v16bf af = cat8(*(const v8bf*)pa, *(const v8bf*)(pa + 8));
    #pragma unroll
    for (int sub = 0; sub < 4; ++sub) {
      const float* p = pw + (size_t)sub * 16 * Dd;
      v8f f0 = *(const v8f*)p;
      v8f f1 = *(const v8f*)(p + 8);
      v16bf bfv;
      #pragma unroll
      for (int j = 0; j < 8; ++j) { bfv[j] = (bf16)f0[j]; bfv[j + 8] = (bf16)f1[j]; }
      acc[sub] = wmma_bf16(af, bfv, acc[sub]);
    }
    pa += 32; pw += 32;
    __builtin_prefetch(pw, 0, 1);
  }
  #pragma unroll
  for (int r = 0; r < 8; ++r) {
    int rowb = r + 8 * g;
    if (rowb < Bb) {
      #pragma unroll
      for (int sub = 0; sub < 4; ++sub) {
        int col = n0 + sub * 16 + m;
        out[(size_t)rowb * Vv + col] = acc[sub][r] + bo[col];
      }
    }
  }
}

extern "C" void kernel_launch(void* const* d_in, const int* in_sizes, int n_in,
                              void* d_out, int out_size, void* d_ws, size_t ws_size,
                              hipStream_t stream) {
  (void)in_sizes; (void)n_in; (void)out_size; (void)ws_size;
  const int*   x      = (const int*)  d_in[0];
  const float* emb    = (const float*)d_in[1];
  const float* norm_w = (const float*)d_in[2];
  const float* norm_b = (const float*)d_in[3];
  const float* b_mat  = (const float*)d_in[4];
  const float* c_mat  = (const float*)d_in[5];
  const float* d_w    = (const float*)d_in[6];
  const float* d_b    = (const float*)d_in[7];
  const float* a_log  = (const float*)d_in[8];
  const float* dt_log = (const float*)d_in[9];
  const float* fn_w   = (const float*)d_in[10];
  const float* fn_b   = (const float*)d_in[11];
  const float* out_w  = (const float*)d_in[12];
  const float* out_b  = (const float*)d_in[13];
  float* logits = (float*)d_out;

  char* wp = (char*)d_ws;
  auto alloc = [&](size_t bytes) -> char* {
    char* p = wp; wp += (bytes + 255) & ~(size_t)255; return p;
  };
  float* h    = (float*)alloc((size_t)NROWS * Dd * 4);       // residual stream
  float* vbuf = (float*)alloc((size_t)NROWS * Dd * 4);       // v / local scan (in place)
  bf16*  u16  = (bf16*) alloc((size_t)NROWS * Dd * 2);       // LN output, bf16
  bf16*  st16 = (bf16*) alloc((size_t)NROWS * Dd * 2);       // scan states, bf16
  float* cl   = (float*)alloc((size_t)Bb * NC * Dd * 4);     // chunk last / carry-in
  float* abar = (float*)alloc((size_t)Ll * Dd * 4);
  float* scl  = (float*)alloc((size_t)Ll * Dd * 4);
  bf16*  wb   = (bf16*) alloc((size_t)Ll * Dd * Dd * 2);
  bf16*  wc   = (bf16*) alloc((size_t)Ll * Dd * Dd * 2);
  bf16*  wd   = (bf16*) alloc((size_t)Ll * Dd * Dd * 2);
  bf16*  hf   = (bf16*) alloc((size_t)16 * Dd * 2);

  k_embed <<<(NROWS * (Dd / 4)) / 256, 256, 0, stream>>>(x, emb, h);
  k_params<<<(Ll * Dd + 255) / 256, 256, 0, stream>>>(a_log, dt_log, abar, scl);
  k_cvt   <<<(Ll * Dd * Dd) / 256, 256, 0, stream>>>(b_mat, c_mat, d_w, scl, wb, wc, wd);

  const int gemmBlocks = (NROWS / 16) * (Dd / 64) / 8;       // 2048 blocks x 8 waves
  for (int l = 0; l < Ll; ++l) {
    size_t wOff = (size_t)l * Dd * Dd;
    k_ln        <<<NROWS / 8, 256, 0, stream>>>(h, norm_w + l * Dd, norm_b + l * Dd, u16);
    k_gemm_v    <<<gemmBlocks, 256, 0, stream>>>(u16, wb + wOff, vbuf);
    k_scan_local<<<(Bb * NC * Dd) / 256, 256, 0, stream>>>(vbuf, abar + l * Dd, cl);
    k_scan_carry<<<(Bb * Dd) / 256, 256, 0, stream>>>(cl, abar + l * Dd);
    k_scan_fix  <<<(Bb * NC * Dd) / 256, 256, 0, stream>>>(vbuf, cl, abar + l * Dd, st16);
    k_gemm_y    <<<gemmBlocks, 256, 0, stream>>>(st16, u16, wc + wOff, wd + wOff,
                                                 d_b + l * Dd, h);
  }
  k_final_ln<<<1, 512, 0, stream>>>(h, fn_w, fn_b, hf);
  k_out<<<(Vv / 64 + 7) / 8, 256, 0, stream>>>(hf, out_w, out_b, logits);
}
